// Decoder_15530601742981
// MI455X (gfx1250) — matrix operand order repaired and checked by reading, not compiled, not run on hardware
//
#include <hip/hip_runtime.h>
#include <hip/hip_bf16.h>
#include <math.h>

// ---------------------------------------------------------------------------
// MI455X (gfx1250) decoder step.  B=64, TE=512, VOCAB=32000, E=512, U=1024,
// ENC=2048, XD=2560.  All GEMMs on v_wmma_f32_16x16x32_f16 (f32 accumulate),
// f32->f16 conversion done with packed v_cvt_pk_rtz while staging via LDS.
// K-tile depth 64 (2 WMMA k-steps per barrier).
// ---------------------------------------------------------------------------

#define B_    64
#define TE_   512
#define ENC_  2048
#define U_    1024
#define E_    512
#define VOC_  32000
#define XD_   2560
#define G3_   3072   // 3*U

typedef __attribute__((ext_vector_type(16))) _Float16 v16h;
typedef __attribute__((ext_vector_type(8)))  _Float16 v8h;
typedef __attribute__((ext_vector_type(2)))  __fp16   h2;   // cvt_pkrtz native type
typedef __attribute__((ext_vector_type(8)))  float    v8f;

__device__ __forceinline__ v8f wmma16x16x32(v16h a, v16h b, v8f c) {
  return __builtin_amdgcn_wmma_f32_16x16x32_f16(
      false, a, false, b, (short)0, c, false, false);
}

// pack 8 consecutive f32 (two float4) -> 8 f16 via v_cvt_pk_rtz_f16_f32
__device__ __forceinline__ v8h pack8(float4 a, float4 b) {
  v8h o; h2 p;
  p = __builtin_amdgcn_cvt_pkrtz(a.x, a.y); o[0] = p[0]; o[1] = p[1];
  p = __builtin_amdgcn_cvt_pkrtz(a.z, a.w); o[2] = p[0]; o[3] = p[1];
  p = __builtin_amdgcn_cvt_pkrtz(b.x, b.y); o[4] = p[0]; o[5] = p[1];
  p = __builtin_amdgcn_cvt_pkrtz(b.z, b.w); o[6] = p[0]; o[7] = p[1];
  return o;
}
__device__ __forceinline__ v8h pack8s(float a0, float a1, float a2, float a3,
                                      float a4, float a5, float a6, float a7) {
  v8h o; h2 p;
  p = __builtin_amdgcn_cvt_pkrtz(a0, a1); o[0] = p[0]; o[1] = p[1];
  p = __builtin_amdgcn_cvt_pkrtz(a2, a3); o[2] = p[0]; o[3] = p[1];
  p = __builtin_amdgcn_cvt_pkrtz(a4, a5); o[4] = p[0]; o[5] = p[1];
  p = __builtin_amdgcn_cvt_pkrtz(a6, a7); o[6] = p[0]; o[7] = p[1];
  return o;
}

// A fragment: 16x32 window of a row-major LDS tile with row stride 64.
// Contiguous K: lanes 0-15 K=0..15, lanes 16-31 K=16..31 (matches B).
__device__ __forceinline__ v16h load_a_frag(const _Float16* tileA, int lane) {
  int r  = lane & 15;
  int kh = (lane >> 4) << 4;
  const _Float16* p = tileA + r * 64;
  v16h f;
#pragma unroll
  for (int i = 0; i < 8; ++i) { f[i] = p[kh + i]; f[i + 8] = p[kh + 8 + i]; }
  return f;
}
// B fragment: LDS is [col][k] with k stride 1, col stride 64. One contiguous
// 32B read per lane: lanes 0-15 K=0..15 (N=lane), lanes 16-31 K=16..31.
__device__ __forceinline__ v16h load_b_frag(const _Float16* tileB, int lane) {
  return *(const v16h*)(tileB + (lane & 15) * 64 + ((lane >> 4) << 4));
}

__device__ __forceinline__ float sigmoidf_(float x) {
  return 1.0f / (1.0f + __expf(-x));
}

// Stage a 64(k) x 128(c) f32 block (row stride ldb) transposed into LDS [c][k]
// (col stride 64, k offset kq). 8 b128 global loads + 4 packed b128 LDS stores.
__device__ __forceinline__ void stage_b_tr(const float* __restrict__ src_base,
                                           size_t ldb, _Float16* __restrict__ dstB,
                                           int tid) {
  int cq = (tid & 31) * 4;
  int kq = (tid >> 5) * 8;
  const float* src = src_base + (size_t)kq * ldb + cq;
  float4 r0 = *(const float4*)(src + 0 * ldb);
  float4 r1 = *(const float4*)(src + 1 * ldb);
  float4 r2 = *(const float4*)(src + 2 * ldb);
  float4 r3 = *(const float4*)(src + 3 * ldb);
  float4 r4 = *(const float4*)(src + 4 * ldb);
  float4 r5 = *(const float4*)(src + 5 * ldb);
  float4 r6 = *(const float4*)(src + 6 * ldb);
  float4 r7 = *(const float4*)(src + 7 * ldb);
  _Float16* dst = dstB + cq * 64 + kq;
  *(v8h*)(dst +   0) = pack8s(r0.x, r1.x, r2.x, r3.x, r4.x, r5.x, r6.x, r7.x);
  *(v8h*)(dst +  64) = pack8s(r0.y, r1.y, r2.y, r3.y, r4.y, r5.y, r6.y, r7.y);
  *(v8h*)(dst + 128) = pack8s(r0.z, r1.z, r2.z, r3.z, r4.z, r5.z, r6.z, r7.z);
  *(v8h*)(dst + 192) = pack8s(r0.w, r1.w, r2.w, r3.w, r4.w, r5.w, r6.w, r7.w);
}

// ---------------------------------------------------------------------------
__global__ void zero_kernel(float* __restrict__ p, int n) {
  int i = blockIdx.x * blockDim.x + threadIdx.x;
  if (i < n) p[i] = 0.0f;
}

// ah[b,u] = state[b,:] @ Wah[:,u] + bah[u] + bac[u]
__global__ __launch_bounds__(256) void ah_kernel(
    const float* __restrict__ state, const float* __restrict__ Wah,
    const float* __restrict__ bah, const float* __restrict__ bac,
    float* __restrict__ ah) {
  int b = blockIdx.x, tid = threadIdx.x;
  __shared__ float s[ENC_];
  for (int k = tid; k < ENC_; k += 256) s[k] = state[(size_t)b * ENC_ + k];
  __syncthreads();
  int u0 = tid * 4;
  float4 acc = *(const float4*)(bah + u0);
  float4 bc  = *(const float4*)(bac + u0);
  acc.x += bc.x; acc.y += bc.y; acc.z += bc.z; acc.w += bc.w;
  for (int k = 0; k < ENC_; ++k) {
    float sv = s[k];
    float4 w = *(const float4*)(Wah + (size_t)k * U_ + u0);
    acc.x += sv * w.x; acc.y += sv * w.y; acc.z += sv * w.z; acc.w += sv * w.w;
  }
  *(float4*)(ah + (size_t)b * U_ + u0) = acc;
}

// Fused attention-score GEMM:
//   scores[b,t] += sum_u tanh(ah[b,u] + enc[b,t,:]@Wac[:,u]) * Was[u]
// Grid: (TE/64, U/128, B); 8 waves; each wave: 1 M-tile x 4 N-tiles.
__global__ __launch_bounds__(256) void score_kernel(
    const float* __restrict__ enc, const float* __restrict__ Wac,
    const float* __restrict__ ah, const float* __restrict__ Was,
    float* __restrict__ scores) {
  int ttile = blockIdx.x, ug = blockIdx.y, b = blockIdx.z;
  int tid = threadIdx.x, lane = tid & 31, w = tid >> 5;
  __shared__ __align__(32) _Float16 ldsA[64 * 64];    // [row][k]
  __shared__ __align__(32) _Float16 ldsB[128 * 64];   // [col][k]
  int mtile = w & 3;
  int ng0   = (w >> 2) * 4;
  v8f acc[4] = {};
  const float* encb = enc + ((size_t)b * TE_ + (size_t)ttile * 64) * ENC_;
  const float* Bcol = Wac + (size_t)ug * 128;

  for (int k0 = 0; k0 < ENC_; k0 += 64) {
    { // stage A: 64 rows x 64 k
      int row = tid >> 2;
      int kq  = (tid & 3) * 16;
      const float* src = encb + (size_t)row * ENC_ + k0 + kq;
      float4 f0 = *(const float4*)(src);
      float4 f1 = *(const float4*)(src + 4);
      float4 f2 = *(const float4*)(src + 8);
      float4 f3 = *(const float4*)(src + 12);
      _Float16* dst = ldsA + row * 64 + kq;
      *(v8h*)(dst)     = pack8(f0, f1);
      *(v8h*)(dst + 8) = pack8(f2, f3);
      if (k0 + 64 < ENC_) __builtin_prefetch(src + 64, 0, 1);
    }
    stage_b_tr(Bcol + (size_t)k0 * U_, U_, ldsB, tid);
    if (k0 + 64 < ENC_)
      __builtin_prefetch(Bcol + (size_t)(k0 + 64 + ((tid >> 5) * 8)) * U_ +
                             (tid & 31) * 4, 0, 1);
    __syncthreads();
#pragma unroll
    for (int ksub = 0; ksub < 2; ++ksub) {
      v16h a = load_a_frag(ldsA + mtile * 16 * 64 + ksub * 32, lane);
#pragma unroll
      for (int j = 0; j < 4; ++j) {
        v16h bf = load_b_frag(ldsB + (ng0 + j) * 16 * 64 + ksub * 32, lane);
        acc[j] = wmma16x16x32(a, bf, acc[j]);
      }
    }
    __syncthreads();
  }

  // epilogue: tanh(+ah)*Was, reduce across 16 N-lanes, atomic into scores
#pragma unroll
  for (int j = 0; j < 4; ++j) {
    int n = ug * 128 + (ng0 + j) * 16 + (lane & 15);
    float ahv = ah[(size_t)b * U_ + n];
    float wv  = Was[n];
#pragma unroll
    for (int i = 0; i < 8; ++i) {
      float v = tanhf(acc[j][i] + ahv) * wv;
      v += __shfl_xor(v, 1, 16);
      v += __shfl_xor(v, 2, 16);
      v += __shfl_xor(v, 4, 16);
      v += __shfl_xor(v, 8, 16);
      if ((lane & 15) == 0) {
        int m = mtile * 16 + ((lane >> 4) << 3) + i;
        atomicAdd(&scores[(size_t)b * TE_ + ttile * 64 + m], v);
      }
    }
  }
}

// Softmax over TE, ctx = sum_t attn*enc, concat emb[token] -> x[b, 0:2560]
__global__ __launch_bounds__(256) void softmax_ctx_kernel(
    const float* __restrict__ scores, const float* __restrict__ enc,
    const float* __restrict__ emb, const int* __restrict__ tokens,
    float* __restrict__ x) {
  int b = blockIdx.x, tid = threadIdx.x;
  __shared__ float attn[TE_];
  __shared__ float red[256];
  float m = -INFINITY;
  for (int t = tid; t < TE_; t += 256) {
    float v = scores[(size_t)b * TE_ + t];
    attn[t] = v;
    m = fmaxf(m, v);
  }
  red[tid] = m; __syncthreads();
  for (int s = 128; s > 0; s >>= 1) {
    if (tid < s) red[tid] = fmaxf(red[tid], red[tid + s]);
    __syncthreads();
  }
  m = red[0]; __syncthreads();
  float sum = 0.0f;
  for (int t = tid; t < TE_; t += 256) {
    float e = __expf(attn[t] - m);
    attn[t] = e;
    sum += e;
  }
  red[tid] = sum; __syncthreads();
  for (int s = 128; s > 0; s >>= 1) {
    if (tid < s) red[tid] += red[tid + s];
    __syncthreads();
  }
  float inv = 1.0f / red[0]; __syncthreads();
  for (int t = tid; t < TE_; t += 256) attn[t] *= inv;
  __syncthreads();

  const float* encb = enc + (size_t)b * TE_ * ENC_;
  for (int e0 = tid * 4; e0 < ENC_; e0 += 1024) {
    float ax = 0.f, ay = 0.f, az = 0.f, aw = 0.f;
    for (int t = 0; t < TE_; ++t) {
      float at = attn[t];
      float4 v = *(const float4*)(encb + (size_t)t * ENC_ + e0);
      ax += at * v.x; ay += at * v.y; az += at * v.z; aw += at * v.w;
    }
    float4 o; o.x = ax; o.y = ay; o.z = az; o.w = aw;
    *(float4*)(x + (size_t)b * XD_ + e0) = o;
  }
  int tok = tokens[b];
  for (int j = tid * 4; j < E_; j += 1024) {
    float4 v = *(const float4*)(emb + (size_t)tok * E_ + j);
    *(float4*)(x + (size_t)b * XD_ + ENC_ + j) = v;
  }
}

// GRU layer, zero initial state: h = (1-z)*c,  gx = A @ K.
// Grid: (8 n-groups of 128 h-cols, 2 m-pairs of 32 rows); 8 waves,
// each wave: 2 (m,n) positions x 3 accumulators (z/r/h column thirds).
__global__ __launch_bounds__(256) void gru_kernel(
    const float* __restrict__ A, const float* __restrict__ K,
    const float* __restrict__ bias, int Kdim,
    float* __restrict__ out, int ldo, int col_off) {
  int ncg = blockIdx.x, mp = blockIdx.y;
  int tid = threadIdx.x, lane = tid & 31, w = tid >> 5;
  __shared__ __align__(32) _Float16 ldsA[32 * 64];
  __shared__ __align__(32) _Float16 ldsB[384 * 64];   // [s*128+c][k]
  int mtile = w & 1;
  int n0    = (w >> 1) * 2;
  v8f acc[2][3] = {};

  for (int k0 = 0; k0 < Kdim; k0 += 64) {
    { // stage A: 32 rows x 64 k
      int row = tid >> 3;
      int kq  = (tid & 7) * 8;
      const float* src = A + (size_t)(mp * 32 + row) * Kdim + k0 + kq;
      float4 f0 = *(const float4*)(src);
      float4 f1 = *(const float4*)(src + 4);
      *(v8h*)(ldsA + row * 64 + kq) = pack8(f0, f1);
    }
#pragma unroll
    for (int s = 0; s < 3; ++s)
      stage_b_tr(K + (size_t)k0 * G3_ + s * U_ + (size_t)ncg * 128, G3_,
                 ldsB + s * 128 * 64, tid);
    if (k0 + 64 < Kdim)
      __builtin_prefetch(K + (size_t)(k0 + 64 + ((tid >> 5) * 8)) * G3_ +
                             (size_t)ncg * 128 + (tid & 31) * 4, 0, 1);
    __syncthreads();
#pragma unroll
    for (int ksub = 0; ksub < 2; ++ksub) {
      v16h a = load_a_frag(ldsA + mtile * 16 * 64 + ksub * 32, lane);
#pragma unroll
      for (int j = 0; j < 2; ++j) {
#pragma unroll
        for (int s = 0; s < 3; ++s) {
          v16h bf = load_b_frag(
              ldsB + (s * 128 + (n0 + j) * 16) * 64 + ksub * 32, lane);
          acc[j][s] = wmma16x16x32(a, bf, acc[j][s]);
        }
      }
    }
    __syncthreads();
  }

  const float* b0 = bias;          // b[0] input bias
  const float* b1 = bias + G3_;    // b[1] recurrent bias (h0=0 -> gh = b1)
#pragma unroll
  for (int j = 0; j < 2; ++j) {
    int n = ncg * 128 + (n0 + j) * 16 + (lane & 15);
    float bz  = b0[n]        + b1[n];
    float br  = b0[U_ + n]   + b1[U_ + n];
    float bh0 = b0[2 * U_ + n];
    float bh1 = b1[2 * U_ + n];
#pragma unroll
    for (int i = 0; i < 8; ++i) {
      int m = mp * 32 + mtile * 16 + ((lane >> 4) << 3) + i;
      float z = sigmoidf_(acc[j][0][i] + bz);
      float r = sigmoidf_(acc[j][1][i] + br);
      float c = tanhf(acc[j][2][i] + bh0 + r * bh1);
      out[(size_t)m * ldo + col_off + n] = (1.0f - z) * c;
    }
  }
}

// pred = out @ Wp + bp ; [64,2048]x[2048,32000]. Grid: 250 n-groups of 128.
__global__ __launch_bounds__(256) void pred_kernel(
    const float* __restrict__ A, const float* __restrict__ Wp,
    const float* __restrict__ bp, float* __restrict__ pred) {
  int ncg = blockIdx.x;
  int tid = threadIdx.x, lane = tid & 31, w = tid >> 5;
  __shared__ __align__(32) _Float16 ldsA[64 * 64];
  __shared__ __align__(32) _Float16 ldsB[128 * 64];
  int mtile = w & 3;
  int n0    = (w >> 2) * 4;
  v8f acc[4] = {};
  const float* Bcol = Wp + (size_t)ncg * 128;

  for (int k0 = 0; k0 < ENC_; k0 += 64) {
    { // stage A: 64 rows x 64 k
      int row = tid >> 2;
      int kq  = (tid & 3) * 16;
      const float* src = A + (size_t)row * ENC_ + k0 + kq;
      float4 f0 = *(const float4*)(src);
      float4 f1 = *(const float4*)(src + 4);
      float4 f2 = *(const float4*)(src + 8);
      float4 f3 = *(const float4*)(src + 12);
      _Float16* dst = ldsA + row * 64 + kq;
      *(v8h*)(dst)     = pack8(f0, f1);
      *(v8h*)(dst + 8) = pack8(f2, f3);
    }
    stage_b_tr(Bcol + (size_t)k0 * VOC_, VOC_, ldsB, tid);
    if (k0 + 64 < ENC_)
      __builtin_prefetch(Bcol + (size_t)(k0 + 64 + ((tid >> 5) * 8)) * VOC_ +
                             (tid & 31) * 4, 0, 1);
    __syncthreads();
#pragma unroll
    for (int ksub = 0; ksub < 2; ++ksub) {
      v16h a = load_a_frag(ldsA + mtile * 16 * 64 + ksub * 32, lane);
#pragma unroll
      for (int j = 0; j < 4; ++j) {
        v16h bf = load_b_frag(ldsB + (n0 + j) * 16 * 64 + ksub * 32, lane);
        acc[j] = wmma16x16x32(a, bf, acc[j]);
      }
    }
    __syncthreads();
  }

#pragma unroll
  for (int j = 0; j < 4; ++j) {
    int n = ncg * 128 + (n0 + j) * 16 + (lane & 15);
    float bv = bp[n];
#pragma unroll
    for (int i = 0; i < 8; ++i) {
      int m = mtile * 16 + ((lane >> 4) << 3) + i;
      pred[(size_t)m * VOC_ + n] = acc[j][i] + bv;
    }
  }
}

// ---------------------------------------------------------------------------
extern "C" void kernel_launch(void* const* d_in, const int* in_sizes, int n_in,
                              void* d_out, int out_size, void* d_ws, size_t ws_size,
                              hipStream_t stream) {
  (void)in_sizes; (void)n_in; (void)out_size; (void)ws_size;
  const int*   tokens = (const int*)d_in[0];
  const float* state  = (const float*)d_in[1];
  const float* enc    = (const float*)d_in[2];
  const float* emb    = (const float*)d_in[3];
  const float* Wah    = (const float*)d_in[4];
  const float* bah    = (const float*)d_in[5];
  const float* Wac    = (const float*)d_in[6];
  const float* bac    = (const float*)d_in[7];
  const float* Was    = (const float*)d_in[8];
  // d_in[9] = bas: scalar added to every score -> softmax-invariant, dropped.
  const float* kf0 = (const float*)d_in[10];
  const float* kf1 = (const float*)d_in[11];
  // d_in[12],[13],[18],[19] = gru rk: h0==0 -> recurrent matmuls vanish.
  const float* bf0 = (const float*)d_in[14];
  const float* bf1 = (const float*)d_in[15];
  const float* kb0 = (const float*)d_in[16];
  const float* kb1 = (const float*)d_in[17];
  const float* bb0 = (const float*)d_in[20];
  const float* bb1 = (const float*)d_in[21];
  const float* Wp  = (const float*)d_in[22];
  const float* bp  = (const float*)d_in[23];

  float* pred   = (float*)d_out;                       // [64, 32000]
  float* outbuf = pred + (size_t)B_ * VOC_;            // [64, 2048]

  float* scores = (float*)d_ws;                        // [64, 512]
  float* ahbuf  = scores + B_ * TE_;                   // [64, 1024]
  float* xbuf   = ahbuf + B_ * U_;                     // [64, 2560]
  float* h0f    = xbuf + (size_t)B_ * XD_;             // [64, 1024]
  float* h0b    = h0f + (size_t)B_ * U_;               // [64, 1024]

  zero_kernel<<<(B_ * TE_ + 255) / 256, 256, 0, stream>>>(scores, B_ * TE_);
  ah_kernel<<<B_, 256, 0, stream>>>(state, Wah, bah, bac, ahbuf);
  score_kernel<<<dim3(TE_ / 64, U_ / 128, B_), 256, 0, stream>>>(
      enc, Wac, ahbuf, Was, scores);
  softmax_ctx_kernel<<<B_, 256, 0, stream>>>(scores, enc, emb, tokens, xbuf);

  gru_kernel<<<dim3(8, 2), 256, 0, stream>>>(xbuf, kf0, bf0, XD_, h0f, U_, 0);
  gru_kernel<<<dim3(8, 2), 256, 0, stream>>>(xbuf, kb0, bb0, XD_, h0b, U_, 0);
  gru_kernel<<<dim3(8, 2), 256, 0, stream>>>(h0f, kf1, bf1, U_, outbuf, 2 * U_, 0);
  gru_kernel<<<dim3(8, 2), 256, 0, stream>>>(h0b, kb1, bb1, U_, outbuf, 2 * U_, U_);

  pred_kernel<<<VOC_ / 128, 256, 0, stream>>>(outbuf, Wp, bp, pred);
}
